// MultiHeadSelfAttention_13855564497125
// MI455X (gfx1250) — matrix operand order repaired and checked by reading, not compiled, not run on hardware
//
#include <hip/hip_runtime.h>
#include <stdint.h>

// ---------------------------------------------------------------------------
// MI455X (gfx1250) multi-head self-attention, bf16 WMMA pipeline + TDM.
// B=2, T=2048, D=1024, H=16, d_k=64.  wave32, V_WMMA_F32_16X16X32_BF16,
// TENSOR_LOAD_TO_LDS double-buffered GEMM staging.
// ---------------------------------------------------------------------------

#define B_SZ 2
#define T_SZ 2048
#define D_SZ 1024
#define H_SZ 16
#define DK_SZ 64
#define M_SZ (B_SZ * T_SZ)  // 4096 rows

typedef __attribute__((ext_vector_type(16))) __bf16 v16bf;
typedef __attribute__((ext_vector_type(8)))  __bf16 v8bf;
typedef __attribute__((ext_vector_type(8)))  float  v8f;
typedef __attribute__((ext_vector_type(4)))  unsigned int v4u;
typedef __attribute__((ext_vector_type(8)))  int v8i;
typedef __attribute__((ext_vector_type(4)))  int v4i;

// fp32 -> bf16, round-to-nearest-even (bit manipulation; no __bf16 arithmetic)
__device__ __forceinline__ __bf16 f2bf(float f) {
  union { float f; uint32_t u; } in; in.f = f;
  uint32_t u = in.u;
  uint32_t r = (u + 0x7FFFu + ((u >> 16) & 1u)) >> 16;
  union { uint16_t s; __bf16 b; } out; out.s = (uint16_t)r;
  return out.b;
}

__device__ __forceinline__ v8f wmma_bf16(v16bf a, v16bf b, v8f c) {
  // (neg_a, A, neg_b, B, c_mod, C, reuse_a, reuse_b)
  return __builtin_amdgcn_wmma_f32_16x16x32_bf16(
      false, a, false, b, (short)0, c, false, false);
}

// A-fragment (16x32, MxK) from row-major A (lda = row stride in elements).
// Lane l: row l%16; K 0..15 (l<16) / 16..31 (l>=16)
// -> two contiguous 16-byte loads.
__device__ __forceinline__ v16bf load_a_frag(const __bf16* A,
                                             int row0, int lda, int kk, int lane) {
  int m = row0 + (lane & 15);
  int koff = (lane >> 4) * 16;
  const __bf16* p = A + (size_t)m * lda + kk + koff;
  union { v16bf v; v8bf h[2]; } u;
  u.h[0] = *reinterpret_cast<const v8bf*>(p);
  u.h[1] = *reinterpret_cast<const v8bf*>(p + 8);
  return u.v;
}

// B-fragment (32x16, KxN) given B-transposed row-major (Bt[n][k], ldb = stride).
// Lane l: fixed n = col0 + l%16, 16 contiguous K values (+16 for upper half).
__device__ __forceinline__ v16bf load_b_frag(const __bf16* Bt,
                                             int col0, int ldb, int kk, int lane) {
  int n = col0 + (lane & 15);
  int koff = kk + (lane >> 4) * 16;
  const __bf16* p = Bt + (size_t)n * ldb + koff;
  union { v16bf v; v8bf h[2]; } u;
  u.h[0] = *reinterpret_cast<const v8bf*>(p);
  u.h[1] = *reinterpret_cast<const v8bf*>(p + 8);
  return u.v;
}

// ---------------------------------------------------------------------------
// Tensor Data Mover: DMA a 2D bf16 tile (rows x cols) from global (row stride
// = stride elems) into LDS at lds_off.  D# packing per CDNA5 ISA ch.8.
// The tile is described as its own tensor (dims == tile dims) so no OOB clip.
// ---------------------------------------------------------------------------
__device__ __forceinline__ void tdm_load_2d_bf16(uint32_t lds_off, const void* gaddr,
                                                 uint32_t rows, uint32_t cols,
                                                 uint64_t stride_elems) {
  uint64_t ga = (uint64_t)(uintptr_t)gaddr;
  v4u g0;
  g0.x = 1u;                                   // count=1, is_restore=0, gather=0
  g0.y = lds_off;                              // LDS byte address
  g0.z = (uint32_t)ga;                         // global_addr[31:0]
  g0.w = (uint32_t)((ga >> 32) & 0x1FFFFFFu) | (2u << 30);  // addr[56:32] | type=2
  v8i g1;
  g1[0] = (int)(1u << 16);                     // wg_mask=0, data_size=1 (2 bytes)
  g1[1] = (int)((cols & 0xFFFFu) << 16);       // tensor_dim0[15:0] at bits 63:48
  g1[2] = (int)((cols >> 16) | ((rows & 0xFFFFu) << 16));   // dim0 hi | dim1 lo
  g1[3] = (int)((rows >> 16) | (cols << 16));  // dim1 hi | tile_dim0 = cols
  g1[4] = (int)(rows & 0xFFFFu);               // tile_dim1 = rows, tile_dim2 = 0
  g1[5] = (int)(uint32_t)stride_elems;         // tensor_dim0_stride[31:0]
  g1[6] = (int)(uint32_t)((stride_elems >> 32) & 0xFFFFu);  // stride hi, dim1_stride=0
  g1[7] = 0;
  v4i gz = {0, 0, 0, 0};
#if __has_include(<hip/amd_detail/amd_gfx1250_TDM.h>)
  v8i gz8 = {0, 0, 0, 0, 0, 0, 0, 0};
  __builtin_amdgcn_tensor_load_to_lds(g0, g1, gz, gz, gz8, 0);   // 6-arg toolchain
#else
  __builtin_amdgcn_tensor_load_to_lds(g0, g1, gz, gz, 0);        // 5-arg ROCm 7.2
#endif
}

__device__ __forceinline__ uint32_t lds_offset_of(const void* p) {
  return (uint32_t)(uintptr_t)p;  // LDS aperture: addr[31:0] is the LDS byte offset
}

// ---------------------------------------------------------------------------
// fp32 -> bf16 conversion
// ---------------------------------------------------------------------------
__global__ __launch_bounds__(256) void k_f32_to_bf16(const float* __restrict__ in,
                                                     __bf16* __restrict__ out, int n) {
  int i = blockIdx.x * 256 + threadIdx.x;
  if (i < n) out[i] = f2bf(in[i]);
}

// ---------------------------------------------------------------------------
// GEMM: C[m,n] = sum_k A[m,k] * Bt[n,k]  (= A @ Bt^T = x @ W^T)
// Block = 8 waves computing a 64x64 tile; per 32-wide k-step the TDM streams
// 64x32 A and B tiles into double-buffered LDS; each wave does a 16x32 slab
// (1 A-frag reused across 2 WMMAs).  MODE:
//   0: store bf16 head-major   out[b][h][t][d]          (Q, K)
//   1: store bf16 transposed   out[b][h][d][t]          (V^T)
//   2: store fp32 row-major    out[m][n]                (final output)
// ---------------------------------------------------------------------------
template <int MODE>
__global__ __launch_bounds__(256) void k_gemm64(const __bf16* __restrict__ A,
                                                const __bf16* __restrict__ Bt,
                                                void* __restrict__ outp,
                                                int M, int N, int K) {
  __shared__ alignas(16) __bf16 lsA[2][64 * 32];   // 4 KB x2
  __shared__ alignas(16) __bf16 lsB[2][64 * 32];   // 4 KB x2

  const int tid = threadIdx.x;
  const int wave = tid >> 5;
  const int lane = tid & 31;
  const int nblk = N >> 6;
  const int bm = blockIdx.x / nblk;
  const int bn = blockIdx.x - bm * nblk;
  const int m_blk = bm << 6, n_blk = bn << 6;
  const int wm = wave & 3;          // 0..3 -> 16-row slab
  const int wn = wave >> 2;         // 0..1 -> 32-col slab

  v8f acc0 = {}, acc1 = {};

  // Prime buffer 0
  if (wave == 0) {
    tdm_load_2d_bf16(lds_offset_of(&lsA[0][0]), A + (size_t)m_blk * K, 64, 32, K);
    tdm_load_2d_bf16(lds_offset_of(&lsB[0][0]), Bt + (size_t)n_blk * K, 64, 32, K);
    __builtin_amdgcn_s_wait_tensorcnt((short)0);
  }
  __syncthreads();

  for (int kk = 0; kk < K; kk += 32) {
    const int cur = (kk >> 5) & 1;
    const int nxt = cur ^ 1;
    if (wave == 0 && kk + 32 < K) {   // stream next tiles while computing
      tdm_load_2d_bf16(lds_offset_of(&lsA[nxt][0]),
                       A + (size_t)m_blk * K + kk + 32, 64, 32, K);
      tdm_load_2d_bf16(lds_offset_of(&lsB[nxt][0]),
                       Bt + (size_t)n_blk * K + kk + 32, 64, 32, K);
    }

    v16bf a  = load_a_frag(&lsA[cur][0], wm * 16, 32, 0, lane);
    v16bf b0 = load_b_frag(&lsB[cur][0], wn * 32, 32, 0, lane);
    v16bf b1 = load_b_frag(&lsB[cur][0], wn * 32 + 16, 32, 0, lane);
    acc0 = wmma_bf16(a, b0, acc0);
    acc1 = wmma_bf16(a, b1, acc1);

    if (wave == 0) __builtin_amdgcn_s_wait_tensorcnt((short)0);
    __syncthreads();
  }

  // Epilogue
  const int rbase = (lane >> 4) * 8;
#pragma unroll
  for (int ai = 0; ai < 2; ++ai) {
    int col = n_blk + wn * 32 + ai * 16 + (lane & 15);
    const v8f& acc = ai ? acc1 : acc0;
#pragma unroll
    for (int r = 0; r < 8; ++r) {
      int row = m_blk + wm * 16 + rbase + r;
      float v = acc[r];
      if (MODE == 2) {
        ((float*)outp)[(size_t)row * N + col] = v;
      } else {
        int b = row >> 11, t = row & (T_SZ - 1);
        int h = col >> 6, d = col & (DK_SZ - 1);
        __bf16* o = (__bf16*)outp;
        if (MODE == 0)
          o[((size_t)((b * H_SZ + h) * T_SZ) + t) * DK_SZ + d] = f2bf(v);
        else
          o[((size_t)((b * H_SZ + h) * DK_SZ) + d) * T_SZ + t] = f2bf(v);
      }
    }
  }
}

// ---------------------------------------------------------------------------
// Flash-attention style causal attention, one wave per 16-row q-tile.
//   Q, K : [B*H, T, 64] bf16 (row-major per head)
//   Vt   : [B*H, 64, T] bf16 (transposed -> contiguous B-fragments)
//   O    : [B, T, 1024] bf16
// ---------------------------------------------------------------------------
__global__ __launch_bounds__(32) void k_attn(const __bf16* __restrict__ Q,
                                             const __bf16* __restrict__ Km,
                                             const __bf16* __restrict__ Vt,
                                             __bf16* __restrict__ O,
                                             float scale) {
  __shared__ alignas(16) __bf16 ptile[16 * 32];  // P-tile relayout (1 wave/block)

  const int lane = threadIdx.x;
  const int hi = lane >> 4;
  const int cl = lane & 15;
  const int qt = blockIdx.x;
  const int bh = blockIdx.y;
  const int q0 = qt << 4;

  const __bf16* Qh = Q + (size_t)bh * T_SZ * DK_SZ;
  const __bf16* Kh = Km + (size_t)bh * T_SZ * DK_SZ;
  const __bf16* Vh = Vt + (size_t)bh * DK_SZ * T_SZ;

  v16bf aq0 = load_a_frag(Qh, q0, DK_SZ, 0, lane);
  v16bf aq1 = load_a_frag(Qh, q0, DK_SZ, 32, lane);

  float mrow[8], lrow[8];
  v8f acc[4] = {{}, {}, {}, {}};
#pragma unroll
  for (int r = 0; r < 8; ++r) { mrow[r] = -1e30f; lrow[r] = 0.f; }

  for (int j = 0; j < q0 + 16; j += 32) {
    v8f s0 = {}, s1 = {};
    {
      v16bf bk = load_b_frag(Kh, j, DK_SZ, 0, lane);
      s0 = wmma_bf16(aq0, bk, s0);
      bk = load_b_frag(Kh, j, DK_SZ, 32, lane);
      s0 = wmma_bf16(aq1, bk, s0);
      bk = load_b_frag(Kh, j + 16, DK_SZ, 0, lane);
      s1 = wmma_bf16(aq0, bk, s1);
      bk = load_b_frag(Kh, j + 16, DK_SZ, 32, lane);
      s1 = wmma_bf16(aq1, bk, s1);
    }

    const int c0 = j + cl, c1 = j + 16 + cl;
    float p0[8], p1[8];
#pragma unroll
    for (int r = 0; r < 8; ++r) {
      int row = q0 + hi * 8 + r;
      float v0 = (c0 <= row) ? s0[r] * scale : -1e30f;  // causal mask
      float v1 = (c1 <= row) ? s1[r] * scale : -1e30f;
      float tmax = fmaxf(v0, v1);
#pragma unroll
      for (int msk = 1; msk < 16; msk <<= 1)
        tmax = fmaxf(tmax, __shfl_xor(tmax, msk, 32));
      float mnew = fmaxf(mrow[r], tmax);
      float fscl = __expf(mrow[r] - mnew);
      float e0 = __expf(v0 - mnew);
      float e1 = __expf(v1 - mnew);
      float rs = e0 + e1;
#pragma unroll
      for (int msk = 1; msk < 16; msk <<= 1)
        rs += __shfl_xor(rs, msk, 32);
      lrow[r] = lrow[r] * fscl + rs;
      mrow[r] = mnew;
#pragma unroll
      for (int nt = 0; nt < 4; ++nt) acc[nt][r] *= fscl;
      p0[r] = e0; p1[r] = e1;
    }

    // Relayout P (C-frag) -> A-frag through LDS
#pragma unroll
    for (int r = 0; r < 8; ++r) {
      int row = hi * 8 + r;
      ptile[row * 32 + cl]      = f2bf(p0[r]);
      ptile[row * 32 + 16 + cl] = f2bf(p1[r]);
    }
    __syncthreads();
    v16bf ap = load_a_frag(ptile, 0, 32, 0, lane);
    __syncthreads();

#pragma unroll
    for (int nt = 0; nt < 4; ++nt) {
      v16bf bv = load_b_frag(Vh, nt * 16, T_SZ, j, lane);
      acc[nt] = wmma_bf16(ap, bv, acc[nt]);
    }
  }

  const int b = bh >> 4, h = bh & (H_SZ - 1);
#pragma unroll
  for (int r = 0; r < 8; ++r) {
    float inv = 1.0f / lrow[r];
    int t = q0 + hi * 8 + r;
#pragma unroll
    for (int nt = 0; nt < 4; ++nt) {
      int col = h * DK_SZ + nt * 16 + cl;
      O[(size_t)(b * T_SZ + t) * D_SZ + col] = f2bf(acc[nt][r] * inv);
    }
  }
}

// ---------------------------------------------------------------------------
// Host-side launch
// ---------------------------------------------------------------------------
extern "C" void kernel_launch(void* const* d_in, const int* in_sizes, int n_in,
                              void* d_out, int out_size, void* d_ws, size_t ws_size,
                              hipStream_t stream) {
  (void)in_sizes; (void)n_in; (void)out_size; (void)ws_size;

  const float* x  = (const float*)d_in[0];
  const float* Wq = (const float*)d_in[1];
  const float* Wk = (const float*)d_in[2];
  const float* Wv = (const float*)d_in[3];
  const float* Wo = (const float*)d_in[4];
  float* out = (float*)d_out;

  char* ws = (char*)d_ws;
  size_t off = 0;
  auto carve = [&](size_t elems) {
    __bf16* p = (__bf16*)(ws + off);
    off = (off + elems * sizeof(__bf16) + 255) & ~(size_t)255;
    return p;
  };
  __bf16* xb  = carve((size_t)M_SZ * D_SZ);
  __bf16* wqb = carve((size_t)D_SZ * D_SZ);
  __bf16* wkb = carve((size_t)D_SZ * D_SZ);
  __bf16* wvb = carve((size_t)D_SZ * D_SZ);
  __bf16* wob = carve((size_t)D_SZ * D_SZ);
  __bf16* Qb  = carve((size_t)M_SZ * D_SZ);  // [B,H,T,64]
  __bf16* Kb  = carve((size_t)M_SZ * D_SZ);  // [B,H,T,64]
  __bf16* Vtb = carve((size_t)M_SZ * D_SZ);  // [B,H,64,T]
  __bf16* Ab  = carve((size_t)M_SZ * D_SZ);  // attention output [B,T,D]

  // 1) fp32 -> bf16 conversions
  {
    int n = M_SZ * D_SZ;
    k_f32_to_bf16<<<(n + 255) / 256, 256, 0, stream>>>(x, xb, n);
    int nw = D_SZ * D_SZ;
    k_f32_to_bf16<<<(nw + 255) / 256, 256, 0, stream>>>(Wq, wqb, nw);
    k_f32_to_bf16<<<(nw + 255) / 256, 256, 0, stream>>>(Wk, wkb, nw);
    k_f32_to_bf16<<<(nw + 255) / 256, 256, 0, stream>>>(Wv, wvb, nw);
    k_f32_to_bf16<<<(nw + 255) / 256, 256, 0, stream>>>(Wo, wob, nw);
  }

  // 2) Q/K/V projections: (4096/64)*(1024/64) = 1024 blocks of 8 waves
  {
    int blocks = (M_SZ / 64) * (D_SZ / 64);
    k_gemm64<0><<<blocks, 256, 0, stream>>>(xb, wqb, Qb, M_SZ, D_SZ, D_SZ);
    k_gemm64<0><<<blocks, 256, 0, stream>>>(xb, wkb, Kb, M_SZ, D_SZ, D_SZ);
    k_gemm64<1><<<blocks, 256, 0, stream>>>(xb, wvb, Vtb, M_SZ, D_SZ, D_SZ);
  }

  // 3) causal flash attention: one wave per (bh, q-tile)
  {
    dim3 grid(T_SZ / 16, B_SZ * H_SZ);
    k_attn<<<grid, 32, 0, stream>>>(Qb, Kb, Vtb, Ab, 0.125f /* 1/sqrt(64) */);
  }

  // 4) output projection -> fp32 d_out
  {
    int blocks = (M_SZ / 64) * (D_SZ / 64);
    k_gemm64<2><<<blocks, 256, 0, stream>>>(Ab, wob, out, M_SZ, D_SZ, D_SZ);
  }
}
